// TransformerBlock_65481071396391
// MI455X (gfx1250) — compile-verified, matrix operand order checked by reading, not run on hardware
//
#include <hip/hip_runtime.h>
#include <hip/hip_bf16.h>

// ---------------------------------------------------------------------------
// Transformer block for MI455X (gfx1250, wave32, WMMA 16x16x32 f16 -> f32),
// with Tensor Data Mover (TDM) async tile DMA into LDS.
//
// Pipeline (f16 matmul operands, f32 accumulate, f32 residual spine):
//   1. transpose_f16: fp32 weights -> f16, stored N-major (N x K).
//   2. layernorm_f16: LN(x) -> f16 activations.
//   3. gemm_wmma<0>: h @ Wqkv + b -> Q[b,h,n,d], K[b,h,n,d], V^T[b,h,d,n]
//   4. flash_attn: online-softmax attention, einsums via v_wmma, K/V via TDM.
//   5. gemm_wmma<1>: attn @ Wout + b + x -> x1 (fp32 residual).
//   6. layernorm_f16: LN(x1) -> f16.
//   7. gemm_wmma<2>: relu(h2 @ W1 + b1) -> f16.
//   8. gemm_wmma<3>: a1 @ W2 + b2 + x1 -> d_out (fp32).
//
// GEMMs: TDM double-buffered 128x32 tiles (64B rows padded to 80B by the TDM
// pad unit), 8 waves x 8 WMMAs per K-step.
// ---------------------------------------------------------------------------

typedef _Float16 v16h __attribute__((ext_vector_type(16)));
typedef _Float16 h8   __attribute__((ext_vector_type(8)));
typedef float    v8f  __attribute__((ext_vector_type(8)));
typedef unsigned int u32x4 __attribute__((ext_vector_type(4)));
typedef int      i32x4 __attribute__((ext_vector_type(4)));
typedef int      i32x8 __attribute__((ext_vector_type(8)));

#define EMBED 1024
#define HEADS 16
#define HEAD_DIM 64
#define FFN 4096
#define SEQ 2048
#define BATCH 2
#define MTOT (BATCH * SEQ)   // 4096 rows

// LDS byte offset of a shared-memory pointer (ptrtoint of addrspace(3)).
__device__ __forceinline__ unsigned lds_addr_of(const void* p) {
    return (unsigned)(unsigned long long)(const __attribute__((address_space(3))) char*)p;
}

// ---------------------------------------------------------------------------
// TDM: async 2-D tile load global->LDS (tracked by TENSORcnt).
//   td0/td1      : tensor dims (elements, elements)
//   s0           : tensor dim0 stride (elements)
//   t0/t1        : tile dims (elements)
//   padi/pada    : LDS pad interval code (2^(c+1) DWORDs) / amount code (c+1 DWORDs)
// D# bit layout per CDNA5 ISA ch.8 (data_size=1 -> 2-byte elements, type=2).
// ---------------------------------------------------------------------------
__device__ __forceinline__ void tdm_load_2d(unsigned lds, const void* gp,
                                            unsigned td0, unsigned td1,
                                            unsigned long long s0,
                                            unsigned t0, unsigned t1,
                                            unsigned padi, unsigned pada) {
    unsigned long long ga = (unsigned long long)gp;
    u32x4 g0;
    g0[0] = 1u;                                    // count=1, user descriptor
    g0[1] = lds;                                   // lds_addr
    g0[2] = (unsigned)ga;                          // global_addr[31:0]
    g0[3] = (unsigned)((ga >> 32) & 0x01FFFFFFull) // global_addr[56:32]
          | (2u << 30);                            // type=2 ("image")
    i32x8 g1;
    g1[0] = (int)((1u << 16)                       // data_size = 2 bytes
                | (1u << 20)                       // pad_enable
                | (padi << 22) | (pada << 25));    // pad interval/amount
    g1[1] = (int)((td0 & 0xFFFFu) << 16);                       // tensor_dim0 lo
    g1[2] = (int)((td0 >> 16) | ((td1 & 0xFFFFu) << 16));       // dim0 hi, dim1 lo
    g1[3] = (int)((td1 >> 16) | (t0 << 16));                    // dim1 hi, tile_dim0
    g1[4] = (int)t1;                                            // tile_dim1 (dim2=0)
    g1[5] = (int)(unsigned)(s0 & 0xFFFFFFFFull);                // dim0_stride lo
    g1[6] = (int)(unsigned)((s0 >> 32) & 0xFFFFull);            // dim0_stride hi
    g1[7] = 0;
    i32x4 z4 = {};
#if __clang_major__ >= 23
    i32x8 z8 = {};
    __builtin_amdgcn_tensor_load_to_lds(g0, g1, z4, z4, z8, 0);
#else
    __builtin_amdgcn_tensor_load_to_lds(g0, g1, z4, z4, 0);
#endif
}

// Load a 16-half A/B WMMA fragment slice: halves {0..7} at p, {8..15} at p+16.
// (CDNA5 16-bit A-matrix layout: lane holds row lane&15, k = 8*(lane>>4) + j
// for j<8, +16 more for j>=8.)
__device__ __forceinline__ v16h frag_ld(const _Float16* p) {
    h8 lo = *(const h8*)(p);
    h8 hi = *(const h8*)(p + 16);
    return __builtin_shufflevector(lo, hi, 0, 1, 2, 3, 4, 5, 6, 7,
                                   8, 9, 10, 11, 12, 13, 14, 15);
}

// ---------------------------------------------------------------------------
// fp32 (K x N) -> f16 transposed (N x K)
// ---------------------------------------------------------------------------
__global__ __launch_bounds__(256) void transpose_f16(
    const float* __restrict__ w, _Float16* __restrict__ wT, int K, int N) {
    long tid = (long)blockIdx.x * 256 + threadIdx.x;
    if (tid < (long)K * N) {
        int k = (int)(tid % K);
        int n = (int)(tid / K);
        wT[tid] = (_Float16)w[(long)k * N + n];
    }
}

// ---------------------------------------------------------------------------
// LayerNorm: one 256-thread block per 1024-wide row, f16 output.
// ---------------------------------------------------------------------------
__global__ __launch_bounds__(256) void layernorm_f16(
    const float* __restrict__ x, const float* __restrict__ g,
    const float* __restrict__ bb, _Float16* __restrict__ y) {
    const int row = blockIdx.x;
    const int tid = threadIdx.x;
    const float* xr = x + (long)row * EMBED;

    float v[4];
    float s = 0.f, ss = 0.f;
#pragma unroll
    for (int i = 0; i < 4; i++) {
        v[i] = xr[tid + 256 * i];
        s += v[i];
        ss += v[i] * v[i];
    }
    for (int m = 1; m < 32; m <<= 1) {
        s  += __shfl_xor(s, m, 32);
        ss += __shfl_xor(ss, m, 32);
    }
    __shared__ float rs[8], rss[8];
    if ((tid & 31) == 0) { rs[tid >> 5] = s; rss[tid >> 5] = ss; }
    __syncthreads();
    s = 0.f; ss = 0.f;
#pragma unroll
    for (int i = 0; i < 8; i++) { s += rs[i]; ss += rss[i]; }
    const float mu  = s * (1.0f / EMBED);
    const float var = ss * (1.0f / EMBED) - mu * mu;
    const float inv = rsqrtf(var + 1e-5f);
#pragma unroll
    for (int i = 0; i < 4; i++) {
        int c = tid + 256 * i;
        y[(long)row * EMBED + c] = (_Float16)((v[i] - mu) * inv * g[c] + bb[c]);
    }
}

// ---------------------------------------------------------------------------
// WMMA GEMM: C(MxN) = A(MxK,f16) * B^T(NxK,f16) + bias, templated epilogue.
// 256 threads = 8 waves, block tile 128x128x32, wave tile 32x64 (8 WMMAs/step).
// A/B tiles DMA'd by the TDM (wave 0 issues, TENSORcnt-waited), double-buffered.
// MODE 0: scatter to Q/K/V^T (f16).  MODE 1: +res -> fp32.  MODE 2: relu f16.
// MODE 3: +res -> fp32 (final output).
// ---------------------------------------------------------------------------
#define BM 128
#define BN 128
#define BK 32
#define LSTR 40   // LDS row stride in halves: 64B data + 16B TDM pad = 80B

template <int MODE>
__global__ __launch_bounds__(256) void gemm_wmma(
    const _Float16* __restrict__ A, const _Float16* __restrict__ Bm,
    const float* __restrict__ bias, const float* __restrict__ res,
    float* __restrict__ out_f32, _Float16* __restrict__ out_h0,
    _Float16* __restrict__ out_h1, _Float16* __restrict__ out_h2,
    int M, int N, int K) {
    __shared__ _Float16 As[2][BM * LSTR];
    __shared__ _Float16 Bs[2][BN * LSTR];

    const int tid   = threadIdx.x;
    const int lane  = tid & 31;
    const int wave  = tid >> 5;
    const int waveM = wave >> 1;  // 0..3 -> 32 rows each
    const int waveN = wave & 1;   // 0..1 -> 64 cols each
    const int m0 = blockIdx.y * BM;
    const int n0 = blockIdx.x * BN;
    const int hi = lane >> 4;
    const int ln = lane & 15;
    const int kbase = hi * 8;

    const unsigned asLds = lds_addr_of(&As[0][0]);
    const unsigned bsLds = lds_addr_of(&Bs[0][0]);
    const unsigned bufBytes = BM * LSTR * 2;

    v8f acc[2][4] = {};

    const int T = K / BK;
    if (wave == 0) {
        tdm_load_2d(asLds, A + (long)m0 * K, K, BM, K, BK, BM, 3, 3);
        tdm_load_2d(bsLds, Bm + (long)n0 * K, K, BN, K, BK, BN, 3, 3);
    }

    for (int t = 0; t < T; ++t) {
        const int cur = t & 1;
        if (wave == 0) __builtin_amdgcn_s_wait_tensorcnt(0);
        __syncthreads();  // tile[cur] ready; all waves done with buf[cur^1]
        if (wave == 0 && t + 1 < T) {
            const long k0n = (long)(t + 1) * BK;
            tdm_load_2d(asLds + (1 - cur) * bufBytes, A + (long)m0 * K + k0n,
                        K, BM, K, BK, BM, 3, 3);
            tdm_load_2d(bsLds + (1 - cur) * bufBytes, Bm + (long)n0 * K + k0n,
                        K, BN, K, BK, BN, 3, 3);
        }

        v16h af[2], bf[4];
#pragma unroll
        for (int tm = 0; tm < 2; tm++)
            af[tm] = frag_ld(As[cur] + (waveM * 32 + tm * 16 + ln) * LSTR + kbase);
#pragma unroll
        for (int tn = 0; tn < 4; tn++)
            bf[tn] = frag_ld(Bs[cur] + (waveN * 64 + tn * 16 + ln) * LSTR + kbase);

#pragma unroll
        for (int tm = 0; tm < 2; tm++)
#pragma unroll
            for (int tn = 0; tn < 4; tn++)
                acc[tm][tn] = __builtin_amdgcn_wmma_f32_16x16x32_f16(
                    false, af[tm], false, bf[tn], (short)0, acc[tm][tn], false,
                    false);
    }

    // Epilogue. C layout: lane holds column n = ln, rows m = r + 8*hi.
#pragma unroll
    for (int tm = 0; tm < 2; tm++) {
#pragma unroll
        for (int tn = 0; tn < 4; tn++) {
            const int n = n0 + waveN * 64 + tn * 16 + ln;
            const float bv = bias[n];
#pragma unroll
            for (int r = 0; r < 8; r++) {
                const int m = m0 + waveM * 32 + tm * 16 + r + 8 * hi;
                const float val = acc[tm][tn][r] + bv;
                if (MODE == 0) {
                    // n -> (section, head, d); m -> (batch, seq)
                    const int sect = n >> 10;
                    const int cc = n & 1023;
                    const int hh = cc >> 6;
                    const int d = cc & 63;
                    const int b = m >> 11;
                    const int nn = m & 2047;
                    if (sect == 0)
                        out_h0[((long)(b * HEADS + hh) * SEQ + nn) * HEAD_DIM + d] =
                            (_Float16)val;
                    else if (sect == 1)
                        out_h1[((long)(b * HEADS + hh) * SEQ + nn) * HEAD_DIM + d] =
                            (_Float16)val;
                    else  // V stored transposed [b,h,d,n]
                        out_h2[((long)(b * HEADS + hh) * HEAD_DIM + d) * SEQ + nn] =
                            (_Float16)val;
                } else if (MODE == 1) {
                    out_f32[(long)m * N + n] = res[(long)m * N + n] + val;
                } else if (MODE == 2) {
                    out_h0[(long)m * N + n] = (_Float16)(val > 0.f ? val : 0.f);
                } else {
                    out_f32[(long)m * N + n] = res[(long)m * N + n] + val;
                }
            }
        }
    }
}

// ---------------------------------------------------------------------------
// Flash attention: grid (SEQ/64, BATCH*HEADS), 128 threads = 4 waves.
// Each wave owns 16 query rows; key blocks of 32 (= WMMA K).
// K/V tiles arrive via TDM (wave 0), S tiles in C layout, row-softmax via
// shfl_xor across 16-lane halves, P -> A-fragment layout via per-wave LDS.
// ---------------------------------------------------------------------------
__global__ __launch_bounds__(128) void flash_attn(
    const _Float16* __restrict__ Q, const _Float16* __restrict__ Kt,
    const _Float16* __restrict__ Vt, _Float16* __restrict__ O) {
    __shared__ _Float16 Ks[32 * 72];      // [key][d], 128B rows + 16B TDM pad
    __shared__ _Float16 Vs[64 * 40];      // [d][key], 64B rows + 16B TDM pad
    __shared__ _Float16 Ps[4 * 16 * 40];  // per-wave P tile (16 x 32)

    const int tid  = threadIdx.x;
    const int lane = tid & 31;
    const int wave = tid >> 5;
    const int hi   = lane >> 4;
    const int ln   = lane & 15;
    const int kbase = hi * 8;
    const int bh = blockIdx.y;
    const long baseQK = (long)bh * SEQ * HEAD_DIM;

    const unsigned ksLds = lds_addr_of(&Ks[0]);
    const unsigned vsLds = lds_addr_of(&Vs[0]);

    // Q fragments: 16 rows x 64 d = two 16x32 A-fragments, straight from global.
    const int qrow = blockIdx.x * 64 + wave * 16 + ln;
    v16h qf[2];
#pragma unroll
    for (int kk = 0; kk < 2; kk++)
        qf[kk] = frag_ld(Q + baseQK + (long)qrow * HEAD_DIM + kk * 32 + kbase);

    v8f o[4] = {};
    float mstat[8], lstat[8];
#pragma unroll
    for (int r = 0; r < 8; r++) { mstat[r] = -3.0e38f; lstat[r] = 0.f; }

    const float scale = 0.125f;  // 1/sqrt(64)
    _Float16* pw = Ps + wave * 16 * 40;

    for (int kb = 0; kb < SEQ; kb += 32) {
        __syncthreads();  // previous iteration's reads of Ks/Vs complete
        if (wave == 0) {
            // K block: 32 rows x 64 elems (128B rows -> pad code 4, +4 DWORDs)
            tdm_load_2d(ksLds, Kt + baseQK + (long)kb * HEAD_DIM,
                        HEAD_DIM, 32, HEAD_DIM, HEAD_DIM, 32, 4, 3);
            // V^T block: 64 rows x 32 elems from [d][n] (stride SEQ)
            tdm_load_2d(vsLds, Vt + (long)bh * HEAD_DIM * SEQ + kb,
                        SEQ, HEAD_DIM, SEQ, 32, HEAD_DIM, 3, 3);
            __builtin_amdgcn_s_wait_tensorcnt(0);
        }
        __syncthreads();  // tiles ready for all waves

        // S tiles: two 16(q) x 16(key) tiles, K-dim = d (2 WMMAs each).
        v8f s[2];
#pragma unroll
        for (int nt = 0; nt < 2; nt++) {
            v8f c = {};
#pragma unroll
            for (int kk = 0; kk < 2; kk++) {
                v16h bfr = frag_ld(Ks + (nt * 16 + ln) * 72 + kk * 32 + kbase);
                c = __builtin_amdgcn_wmma_f32_16x16x32_f16(
                    false, qf[kk], false, bfr, (short)0, c, false, false);
            }
            s[nt] = c;
        }

        // Online softmax per query row (rows r+8*hi; key cols across the
        // 16-lane half -> xor masks 1..8 stay inside the half).
        float pexp[2][8];
#pragma unroll
        for (int r = 0; r < 8; r++) {
            float mx = fmaxf(s[0][r], s[1][r]) * scale;
            for (int msk = 1; msk < 16; msk <<= 1)
                mx = fmaxf(mx, __shfl_xor(mx, msk, 32));
            const float mnew = fmaxf(mstat[r], mx);
            const float corr = __expf(mstat[r] - mnew);
            mstat[r] = mnew;
            const float e0 = __expf(s[0][r] * scale - mnew);
            const float e1 = __expf(s[1][r] * scale - mnew);
            float rsum = e0 + e1;
            for (int msk = 1; msk < 16; msk <<= 1)
                rsum += __shfl_xor(rsum, msk, 32);
            lstat[r] = lstat[r] * corr + rsum;
            pexp[0][r] = e0;
            pexp[1][r] = e1;
#pragma unroll
            for (int t = 0; t < 4; t++) o[t][r] *= corr;
        }

        // P (C layout) -> per-wave LDS tile (q-row major, 32 key cols).
#pragma unroll
        for (int nt = 0; nt < 2; nt++)
#pragma unroll
            for (int r = 0; r < 8; r++)
                pw[(r + 8 * hi) * 40 + nt * 16 + ln] = (_Float16)pexp[nt][r];

        // o += P @ V : A = P fragment (16x32), B = V^T rows (d-major).
        v16h pf = frag_ld(pw + ln * 40 + kbase);
#pragma unroll
        for (int dt = 0; dt < 4; dt++) {
            v16h vf = frag_ld(Vs + (dt * 16 + ln) * 40 + kbase);
            o[dt] = __builtin_amdgcn_wmma_f32_16x16x32_f16(
                false, pf, false, vf, (short)0, o[dt], false, false);
        }
    }

    // Normalize and write attn_out[(b*SEQ+q)*EMBED + h*64 + d] as f16.
    const int b = bh >> 4;
    const int hh = bh & 15;
#pragma unroll
    for (int dt = 0; dt < 4; dt++) {
#pragma unroll
        for (int r = 0; r < 8; r++) {
            const float val = o[dt][r] / lstat[r];
            const int q = blockIdx.x * 64 + wave * 16 + r + 8 * hi;
            const int d = dt * 16 + ln;
            O[((long)(b * SEQ + q)) * EMBED + hh * HEAD_DIM + d] = (_Float16)val;
        }
    }
}

// ---------------------------------------------------------------------------
// Launch
// ---------------------------------------------------------------------------
extern "C" void kernel_launch(void* const* d_in, const int* in_sizes, int n_in,
                              void* d_out, int out_size, void* d_ws,
                              size_t ws_size, hipStream_t stream) {
    const float* x     = (const float*)d_in[0];
    const float* ln1_g = (const float*)d_in[1];
    const float* ln1_b = (const float*)d_in[2];
    const float* ln2_g = (const float*)d_in[3];
    const float* ln2_b = (const float*)d_in[4];
    const float* w_qkv = (const float*)d_in[5];
    const float* b_qkv = (const float*)d_in[6];
    const float* w_out = (const float*)d_in[7];
    const float* b_out = (const float*)d_in[8];
    const float* w1    = (const float*)d_in[9];
    const float* b1    = (const float*)d_in[10];
    const float* w2    = (const float*)d_in[11];
    const float* b2    = (const float*)d_in[12];
    float* out = (float*)d_out;

    // Workspace layout (f16 unless noted).
    _Float16* wqkvT = (_Float16*)d_ws;                  // 3072 x 1024
    _Float16* woutT = wqkvT + (long)3072 * 1024;        // 1024 x 1024
    _Float16* w1T   = woutT + (long)1024 * 1024;        // 4096 x 1024
    _Float16* w2T   = w1T + (long)4096 * 1024;          // 1024 x 4096
    _Float16* hbuf  = w2T + (long)1024 * 4096;          // 4096 x 1024
    _Float16* qb    = hbuf + (long)MTOT * EMBED;        // [b,h,n,d]
    _Float16* kb    = qb + (long)MTOT * EMBED;          // [b,h,n,d]
    _Float16* vtb   = kb + (long)MTOT * EMBED;          // [b,h,d,n]
    _Float16* aob   = vtb + (long)MTOT * EMBED;         // attn out 4096 x 1024
    _Float16* h2b   = aob + (long)MTOT * EMBED;         // 4096 x 1024
    _Float16* a1b   = h2b + (long)MTOT * EMBED;         // 4096 x 4096
    float*    x1    = (float*)(a1b + (long)MTOT * FFN); // 4096 x 1024 fp32

    // 1. Weight conversion/transpose (fp32 KxN -> f16 NxK).
    transpose_f16<<<(1024 * 3072) / 256, 256, 0, stream>>>(w_qkv, wqkvT, 1024, 3072);
    transpose_f16<<<(1024 * 1024) / 256, 256, 0, stream>>>(w_out, woutT, 1024, 1024);
    transpose_f16<<<(1024 * 4096) / 256, 256, 0, stream>>>(w1, w1T, 1024, 4096);
    transpose_f16<<<(4096 * 1024) / 256, 256, 0, stream>>>(w2, w2T, 4096, 1024);

    // 2. LN1
    layernorm_f16<<<MTOT, 256, 0, stream>>>(x, ln1_g, ln1_b, hbuf);

    // 3. QKV GEMM + scatter
    gemm_wmma<0><<<dim3(3072 / BN, MTOT / BM), 256, 0, stream>>>(
        hbuf, wqkvT, b_qkv, nullptr, nullptr, qb, kb, vtb, MTOT, 3072, 1024);

    // 4. Attention
    flash_attn<<<dim3(SEQ / 64, BATCH * HEADS), 128, 0, stream>>>(qb, kb, vtb, aob);

    // 5. Out projection + residual -> x1 (fp32)
    gemm_wmma<1><<<dim3(1024 / BN, MTOT / BM), 256, 0, stream>>>(
        aob, woutT, b_out, x, x1, nullptr, nullptr, nullptr, MTOT, 1024, 1024);

    // 6. LN2
    layernorm_f16<<<MTOT, 256, 0, stream>>>(x1, ln2_g, ln2_b, h2b);

    // 7. FFN up + relu
    gemm_wmma<2><<<dim3(4096 / BN, MTOT / BM), 256, 0, stream>>>(
        h2b, w1T, b1, nullptr, nullptr, a1b, nullptr, nullptr, MTOT, 4096, 1024);

    // 8. FFN down + residual -> d_out
    gemm_wmma<3><<<dim3(1024 / BN, MTOT / BM), 256, 0, stream>>>(
        a1b, w2T, b2, x1, out, nullptr, nullptr, nullptr, MTOT, 1024, 4096);
}
